// RNNModel_4629974745901
// MI455X (gfx1250) — matrix operand order repaired and checked by reading, not compiled, not run on hardware
//
#include <hip/hip_runtime.h>

typedef __attribute__((ext_vector_type(16))) _Float16 v16h;
typedef __attribute__((ext_vector_type(8)))  _Float16 v8h;
typedef __attribute__((ext_vector_type(8)))  float    v8f;
typedef __attribute__((ext_vector_type(4)))  float    v4f;

#define T_STEPS 750
#define BATCH   1024
#define HID     100
#define KPAD    128   // augmented K: 100 hidden + 4 input + 1 bias -> pad to 128
#define NPAD    128   // padded output hidden dim
#define ALPHA     0.01f   // DT / TAU
#define NOISE_STD 0.1f

// Noise tile per WG per step: 16 rows x 100 f32 = 6400 B contiguous = 400 x 16B chunks.
#define NCHUNKS 400

// One workgroup = 8 waves (wave32) = 256 threads, owns a batch tile of 16.
// Wave w computes output hidden columns [16w, 16w+16). Per step:
//   D = A * Baug   with A = [tanh(x) | u_t | 1], Baug = [J^T ; Bmat ; c_x]
// as 4 v_wmma_f32_16x16x32_f16 (two independent accumulate chains), then the
// leaky elementwise update. Noise is streamed one step ahead via
// global_load_async_to_lds_b128 (ASYNCcnt) into a double-buffered LDS tile.
__global__ __launch_bounds__(256) void rnn_scan_kernel(
    const float* __restrict__ input_seq, // [T, B, 4]
    const float* __restrict__ noise,     // [T, B, H]
    const float* __restrict__ J_w,       // [H, H]
    const float* __restrict__ Bmat,      // [4, H]
    const float* __restrict__ c_x,       // [H]
    const float* __restrict__ Wout_w,    // [1, H]
    const float* __restrict__ Wout_b,    // [1]
    float* __restrict__ out)             // [B]
{
  __shared__ _Float16 ldsJ[NPAD * KPAD];                 // ldsJ[n*KPAD+k] = Baug[k,n]
  __shared__ _Float16 ldsR[2][16 * KPAD];                // A tiles (double buffer)
  __shared__ __align__(16) float ldsN[2][16 * HID];      // noise tiles (double buffer)
  __shared__ float    ldsOut[16];

  const int tid  = threadIdx.x;
  const int wave = tid >> 5;
  const int lane = tid & 31;
  const int lh   = lane >> 4;   // lane half (0/1)
  const int ln   = lane & 15;
  const int b0   = blockIdx.x * 16;

  // ---- async-copy issue helper: noise[tt] tile -> ldsN[bb] (16B/lane chunks) ----
  const unsigned int ldsNoff0 = (unsigned int)(uintptr_t)&ldsN[0][0];
  const unsigned int ldsNoff1 = (unsigned int)(uintptr_t)&ldsN[1][0];
#define ISSUE_NOISE_ASYNC(tt, bb)                                               \
  do {                                                                          \
    const unsigned long long gbase =                                            \
        (unsigned long long)(uintptr_t)&noise[((size_t)(tt)*BATCH + b0) * HID]; \
    const unsigned int lbase = (bb) ? ldsNoff1 : ldsNoff0;                      \
    _Pragma("unroll")                                                           \
    for (int c = tid; c < NCHUNKS; c += 256) {                                  \
      unsigned int       ld = lbase + (unsigned int)c * 16u;                    \
      unsigned long long ga = gbase + (unsigned long long)c * 16ull;            \
      asm volatile("global_load_async_to_lds_b128 %0, %1, off"                  \
                   :: "v"(ld), "v"(ga) : "memory");                             \
    }                                                                           \
  } while (0)

  // kick off step-0 noise while we build the LDS tables
  ISSUE_NOISE_ASYNC(0, 0);

  // ---- build augmented B matrix (J^T + Bmat rows + c_x row) in LDS, f16 ----
  for (int idx = tid; idx < NPAD * KPAD; idx += 256) {
    const int n = idx >> 7;          // output hidden index
    const int k = idx & (KPAD - 1);  // contraction index
    float v = 0.f;
    if (n < HID) {
      if (k < HID)           v = J_w[n * HID + k];       // B[k,n] = J[n,k]
      else if (k < HID + 4)  v = Bmat[(k - HID) * HID + n];
      else if (k == HID + 4) v = c_x[n];
    }
    ldsJ[idx] = (_Float16)v;
  }
  // zero both A buffers (r_0 = tanh(0) = 0), set bias column
  for (int idx = tid; idx < 16 * KPAD; idx += 256) {
    ldsR[0][idx] = (_Float16)0.f;
    ldsR[1][idx] = (_Float16)0.f;
  }
  if (tid < 16) {
    ldsR[0][tid * KPAD + HID + 4] = (_Float16)1.f;
    ldsR[1][tid * KPAD + HID + 4] = (_Float16)1.f;
  }
  // stage u_0 into buffer 0 (wave 7's real columns are all padding)
  if (wave == 7 && lane < 16) {
    v4f u = *(const v4f*)&input_seq[(size_t)(0 * BATCH + b0 + lane) * 4];
    _Float16* p = &ldsR[0][lane * KPAD + HID];
    p[0] = (_Float16)u.x; p[1] = (_Float16)u.y;
    p[2] = (_Float16)u.z; p[3] = (_Float16)u.w;
  }
  __syncthreads();

  // ---- resident B fragments for this wave's N-tile (contiguous k = kt*32 + 16*lh + s, same as A) ----
  const int N0 = wave * 16;
  v16h bfrag[4];
  {
    const _Float16* bp = &ldsJ[(N0 + ln) * KPAD + lh * 16];
#pragma unroll
    for (int kt = 0; kt < 4; ++kt)
      bfrag[kt] = *(const v16h*)(bp + kt * 32);
  }

  const int   hcol   = N0 + ln;                  // this lane's hidden column
  const bool  hvalid = (hcol < HID);
  const int   hclamp = hvalid ? hcol : 0;        // branch-free noise column
  const float nscale = hvalid ? (ALPHA * NOISE_STD) : 0.f;
  const float wh     = hvalid ? Wout_w[hcol] : 0.f;

  // x state in registers, C-fragment layout: xreg[j] = x[m = j + 8*lh, hcol]
  float xreg[8];
#pragma unroll
  for (int j = 0; j < 8; ++j) xreg[j] = 0.f;

  asm volatile("s_wait_asynccnt 0x0" ::: "memory");  // noise tile 0 landed
  __syncthreads();

  for (int t = 0; t < T_STEPS; ++t) {
    const int cur = t & 1;

    // stream next step's noise into the other LDS buffer (overlaps compute)
    if (t + 1 < T_STEPS) ISSUE_NOISE_ASYNC(t + 1, cur ^ 1);

    // ---- load all 4 A fragments, then two independent WMMA chains ----
    v16h afr[4];
    {
      const _Float16* ap = ldsR[cur] + ln * KPAD + lh * 16;
#pragma unroll
      for (int kt = 0; kt < 4; ++kt) {
        v8h lo = *(const v8h*)(ap + kt * 32);       // K = kt*32 + 16*lh     .. +8
        v8h hi = *(const v8h*)(ap + kt * 32 + 8);   // K = kt*32 + 16*lh + 8 .. +8
#pragma unroll
        for (int i = 0; i < 8; ++i) { afr[kt][i] = lo[i]; afr[kt][i + 8] = hi[i]; }
      }
    }
    v8f acc0 = {}, acc1 = {};
    acc0 = __builtin_amdgcn_wmma_f32_16x16x32_f16(false, afr[0], false, bfrag[0], (short)0, acc0, false, false);
    acc1 = __builtin_amdgcn_wmma_f32_16x16x32_f16(false, afr[1], false, bfrag[1], (short)0, acc1, false, false);
    acc0 = __builtin_amdgcn_wmma_f32_16x16x32_f16(false, afr[2], false, bfrag[2], (short)0, acc0, false, false);
    acc1 = __builtin_amdgcn_wmma_f32_16x16x32_f16(false, afr[3], false, bfrag[3], (short)0, acc1, false, false);

    // ---- elementwise leaky update: x += alpha*(-x + D) + alpha*0.1*noise ----
    const float* nt = &ldsN[cur][0];
#pragma unroll
    for (int j = 0; j < 8; ++j) {
      const int m = j + 8 * lh;
      const float nz = nt[m * HID + hclamp];       // LDS, branch-free
      float xj = xreg[j];
      xj += ALPHA * ((acc0[j] + acc1[j]) - xj) + nscale * nz;
      xreg[j] = xj;
    }

    if (t + 1 < T_STEPS) {
      _Float16* Rn = ldsR[cur ^ 1];
      // write r = tanh(x) (f16) into next A buffer (only real columns)
      if (hvalid) {
#pragma unroll
        for (int j = 0; j < 8; ++j) {
          const int m = j + 8 * lh;
          Rn[m * KPAD + hcol] = (_Float16)tanhf(xreg[j]);
        }
      }
      // stage u_{t+1}
      if (wave == 7 && lane < 16) {
        v4f u = *(const v4f*)&input_seq[(size_t)((t + 1) * BATCH + b0 + lane) * 4];
        _Float16* p = &Rn[lane * KPAD + HID];
        p[0] = (_Float16)u.x; p[1] = (_Float16)u.y;
        p[2] = (_Float16)u.z; p[3] = (_Float16)u.w;
      }
    }
    // next-step noise must have landed before anyone reads it after the barrier
    asm volatile("s_wait_asynccnt 0x0" ::: "memory");
    __syncthreads();
  }

  // ---- readout: out[b] = tanh(x_T) . Wout + Wout_b ----
  if (tid < 16) ldsOut[tid] = 0.f;
  __syncthreads();
#pragma unroll
  for (int j = 0; j < 8; ++j) {
    const int m = j + 8 * lh;
    atomicAdd(&ldsOut[m], tanhf(xreg[j]) * wh);   // ds_add_f32
  }
  __syncthreads();
  if (tid < 16) out[b0 + tid] = ldsOut[tid] + Wout_b[0];
#undef ISSUE_NOISE_ASYNC
}

extern "C" void kernel_launch(void* const* d_in, const int* in_sizes, int n_in,
                              void* d_out, int out_size, void* d_ws, size_t ws_size,
                              hipStream_t stream) {
  const float* input_seq = (const float*)d_in[0];
  const float* noise     = (const float*)d_in[1];
  const float* J_w       = (const float*)d_in[2];
  const float* Bmat      = (const float*)d_in[3];
  const float* c_x       = (const float*)d_in[4];
  const float* Wout_w    = (const float*)d_in[5];
  const float* Wout_b    = (const float*)d_in[6];
  float* out = (float*)d_out;
  (void)in_sizes; (void)n_in; (void)out_size; (void)d_ws; (void)ws_size;

  dim3 grid(BATCH / 16);   // 64 workgroups, one batch tile of 16 each
  dim3 block(256);         // 8 wave32 waves; wave w owns N-tile w
  hipLaunchKernelGGL(rnn_scan_kernel, grid, block, 0, stream,
                     input_seq, noise, J_w, Bmat, c_x, Wout_w, Wout_b, out);
}
